// VarianceAdaptor_38646115729695
// MI455X (gfx1250) — compile-verified, matrix operand order checked by reading, not run on hardware
//
#include <hip/hip_runtime.h>
#include <hip/hip_bf16.h>

typedef __attribute__((ext_vector_type(16))) _Float16 v16h;
typedef __attribute__((ext_vector_type(8)))  _Float16 v8h;
typedef __attribute__((ext_vector_type(8)))  float    v8f;

#define BB 16
#define TT 512
#define FF 384
#define ML 4096
#define KB_PER_TAP 12                 // 384 / 32
#define OT_TILES   24                 // 384 / 16
#define FRAG_HALVES 512               // 32x16 f16 fragment
#define LAYER_FRAG_HALVES (3*KB_PER_TAP*OT_TILES*FRAG_HALVES)   // 442368
#define MROWS 32                      // rows (time steps) per block

// ---------------------------------------------------------------------------
// 1) durations -> cumsum, searchsorted index map, mel_len; pass-through outs
// ---------------------------------------------------------------------------
__global__ void dur_index_kernel(const int* __restrict__ dur,
                                 int* __restrict__ idx, int* __restrict__ melLen,
                                 float* __restrict__ sd_out, float* __restrict__ mln_out,
                                 float* __restrict__ mm_out)
{
    __shared__ int s[TT];
    int b = blockIdx.x, tid = threadIdx.x;
    int v = dur[b * TT + tid];
    s[tid] = v;
    __syncthreads();
    for (int off = 1; off < TT; off <<= 1) {
        int p = (tid >= off) ? s[tid - off] : 0;
        __syncthreads();
        s[tid] += p;
        __syncthreads();
    }
    int mlen = s[TT - 1];
    if (tid == 0) { melLen[b] = mlen; mln_out[b] = (float)mlen; }
    sd_out[b * TT + tid] = (float)v;

    for (int j = 0; j < ML / TT; j++) {
        int t = tid + j * TT;
        // searchsorted(cum, t, side='right')  ==  first index with cum[i] > t
        int lo = 0, hi = TT;
        while (lo < hi) { int mid = (lo + hi) >> 1; if (s[mid] <= t) lo = mid + 1; else hi = mid; }
        idx[b * ML + t] = (lo < TT - 1) ? lo : (TT - 1);
        mm_out[b * ML + t] = (t >= mlen) ? 1.0f : 0.0f;
    }
}

// ---------------------------------------------------------------------------
// 2) f32 -> f16 (duration-predictor input)
// ---------------------------------------------------------------------------
__global__ void f32_to_f16_kernel(const float* __restrict__ in, _Float16* __restrict__ out, int n)
{
    int base = (blockIdx.x * blockDim.x + threadIdx.x) * 8;
    if (base >= n) return;
#pragma unroll
    for (int j = 0; j < 8; j++) out[base + j] = (_Float16)in[base + j];
}

// ---------------------------------------------------------------------------
// 3) length-regulated predictor input: mask*(x[idx] + sig[idx]*c1w + c1b) -> f16
// ---------------------------------------------------------------------------
__global__ void build_var_input_kernel(const float* __restrict__ x,
                                       const float* __restrict__ sig,   // (B,T,1)
                                       const float* __restrict__ c1w,   // (F)
                                       const float* __restrict__ c1b,   // (F)
                                       const int* __restrict__ idx,
                                       const int* __restrict__ melLen,
                                       _Float16* __restrict__ out)
{
    long i = (long)(blockIdx.x * blockDim.x + threadIdx.x) * 8;
    int  f0 = (int)(i % FF);
    long row = i / FF;                 // b*ML + t
    int  b = (int)(row / ML);
    int  t = (int)(row % ML);
    if (t >= melLen[b]) {
#pragma unroll
        for (int j = 0; j < 8; j++) out[i + j] = (_Float16)0.0f;
        return;
    }
    int   j0 = idx[row];
    float sc = sig[b * TT + j0];
    const float* xp = x + ((long)(b * TT + j0)) * FF + f0;
#pragma unroll
    for (int j = 0; j < 8; j++)
        out[i + j] = (_Float16)(xp[j] + sc * c1w[f0 + j] + c1b[f0 + j]);
}

// ---------------------------------------------------------------------------
// 4) repack conv weights (3,F,F) f32 -> per-fragment f16 WMMA-B layout
//    fragment id = (k*12 + kb)*24 + otile ; lane holds 16 halves (32B)
// ---------------------------------------------------------------------------
__global__ void repack_w_kernel(const float* __restrict__ w, _Float16* __restrict__ frag)
{
    int gid = blockIdx.x * blockDim.x + threadIdx.x;
    int lane = gid & 31;
    int fid  = gid >> 5;
    if (fid >= 3 * KB_PER_TAP * OT_TILES) return;
    int ot = fid % OT_TILES;
    int kk = fid / OT_TILES;
    int kb = kk % KB_PER_TAP;
    int k  = kk / KB_PER_TAP;
    int n  = lane & 15, kg = lane >> 4;
    _Float16* o = frag + (long)fid * FRAG_HALVES + lane * 16;
#pragma unroll
    for (int j = 0; j < 16; j++) {
        int K  = (j & 7) + ((j >> 3) << 4) + kg * 8;   // 16-bit operand VGPR layout
        int ci = kb * 32 + K;
        o[j] = (_Float16)w[((long)k * FF + ci) * FF + ot * 16 + n];
    }
}

// ---------------------------------------------------------------------------
// 5) conv1d(K=3) + bias + ReLU + LayerNorm, via v_wmma_f32_16x16x32_f16.
//    Block = 32 time rows x 384 outputs; 8 waves x (2 row-tiles x 3 col-tiles).
//    Each B fragment feeds 2 WMMAs -> halves per-layer weight L2 traffic.
//    LDS: A tile (34x384 f16, 26 KB) unioned with LN staging (32x384 f32, 48 KB)
//    -- A tile is dead after the K-loop, separated by a barrier.
// ---------------------------------------------------------------------------
__global__ __launch_bounds__(256)
void conv_ln_wmma_kernel(const _Float16* __restrict__ in,
                         const _Float16* __restrict__ wfrag,
                         const float* __restrict__ cb,
                         const float* __restrict__ cg,
                         const float* __restrict__ cbe,
                         _Float16* __restrict__ out,
                         int Tlen)
{
    __shared__ __align__(16) char smem[MROWS * FF * 4];   // 49152 B
    _Float16* sA = (_Float16*)smem;                       // 34 x 384 halves (26112 B)
    float*    sY = (float*)smem;                          // 32 x 384 floats

    int tb = blockIdx.x % (Tlen / MROWS);
    int b  = blockIdx.x / (Tlen / MROWS);
    int t0 = tb * MROWS;
    int tid = threadIdx.x;

    // stage A tile: rows t0-1 .. t0+32 (34 rows x 384 halves), zero-padded
    for (int c = tid; c < 34 * 48; c += 256) {
        int rr = c / 48, f8 = (c % 48) * 8;
        int t = t0 - 1 + rr;
        v8h val = {};
        if (t >= 0 && t < Tlen)
            val = *(const v8h*)(in + ((long)(b * Tlen + t)) * FF + f8);
        *(v8h*)(sA + rr * FF + f8) = val;
    }
    __syncthreads();

    int lane = tid & 31;
    int wave = tid >> 5;
    int m = lane & 15, kg = lane >> 4;

    v8f c0a = {}, c1a = {}, c2a = {};     // rows t0+m    (tile A)
    v8f c0b = {}, c1b_ = {}, c2b = {};    // rows t0+16+m (tile B)
    for (int k = 0; k < 3; k++) {
        for (int kb = 0; kb < KB_PER_TAP; kb++) {
            int rowA = m + k;                      // sA row for x[t0+m+k-1]
            int rowB = rowA + 16;                  // sA row for x[t0+16+m+k-1]
            int cbase = kb * 32 + kg * 8;

            v8h alo = *(const v8h*)(sA + rowA * FF + cbase);
            v8h ahi = *(const v8h*)(sA + rowA * FF + cbase + 16);
            v8h blo = *(const v8h*)(sA + rowB * FF + cbase);
            v8h bhi = *(const v8h*)(sA + rowB * FF + cbase + 16);
            v16h aA, aB;
#pragma unroll
            for (int j = 0; j < 8; j++) {
                aA[j] = alo[j]; aA[j + 8] = ahi[j];
                aB[j] = blo[j]; aB[j + 8] = bhi[j];
            }

            long fb = ((long)((k * KB_PER_TAP + kb) * OT_TILES + wave * 3)) * FRAG_HALVES + lane * 16;
            v16h w0 = *(const v16h*)(wfrag + fb);
            v16h w1 = *(const v16h*)(wfrag + fb + FRAG_HALVES);
            v16h w2 = *(const v16h*)(wfrag + fb + 2 * FRAG_HALVES);
            if (kb + 1 < KB_PER_TAP)
                __builtin_prefetch(wfrag + fb + (long)OT_TILES * FRAG_HALVES, 0, 3);

            c0a  = __builtin_amdgcn_wmma_f32_16x16x32_f16(false, aA, false, w0, (short)0, c0a,  false, false);
            c0b  = __builtin_amdgcn_wmma_f32_16x16x32_f16(false, aB, false, w0, (short)0, c0b,  false, false);
            c1a  = __builtin_amdgcn_wmma_f32_16x16x32_f16(false, aA, false, w1, (short)0, c1a,  false, false);
            c1b_ = __builtin_amdgcn_wmma_f32_16x16x32_f16(false, aB, false, w1, (short)0, c1b_, false, false);
            c2a  = __builtin_amdgcn_wmma_f32_16x16x32_f16(false, aA, false, w2, (short)0, c2a,  false, false);
            c2b  = __builtin_amdgcn_wmma_f32_16x16x32_f16(false, aB, false, w2, (short)0, c2b,  false, false);
        }
    }

    __syncthreads();   // sA dead; reuse smem as sY

    // bias + relu -> sY  (C layout: VGPR r, lanes0-15: M=r; lanes16-31: M=r+8)
    int n = lane & 15, mh = (lane >> 4) * 8;
    int col0 = wave * 48 + n;
#pragma unroll
    for (int r = 0; r < 8; r++) {
        int mm = r + mh;
        sY[mm * FF + col0]             = fmaxf(c0a[r]  + cb[col0],      0.0f);
        sY[mm * FF + col0 + 16]        = fmaxf(c1a[r]  + cb[col0 + 16], 0.0f);
        sY[mm * FF + col0 + 32]        = fmaxf(c2a[r]  + cb[col0 + 32], 0.0f);
        sY[(mm + 16) * FF + col0]      = fmaxf(c0b[r]  + cb[col0],      0.0f);
        sY[(mm + 16) * FF + col0 + 16] = fmaxf(c1b_[r] + cb[col0 + 16], 0.0f);
        sY[(mm + 16) * FF + col0 + 32] = fmaxf(c2b[r]  + cb[col0 + 32], 0.0f);
    }
    __syncthreads();

    // LayerNorm over F=384 per row: 8 threads per row + shfl_xor(width=8)
    int rowLN = tid >> 3;          // 0..31
    int sub   = tid & 7;
    float s1 = 0.0f, s2 = 0.0f;
    for (int e = sub; e < FF; e += 8) { float v = sY[rowLN * FF + e]; s1 += v; s2 += v * v; }
#pragma unroll
    for (int off = 4; off >= 1; off >>= 1) {
        s1 += __shfl_xor(s1, off, 8);
        s2 += __shfl_xor(s2, off, 8);
    }
    float mean = s1 * (1.0f / FF);
    float var  = s2 * (1.0f / FF) - mean * mean;
    float inv  = rsqrtf(var + 1e-5f);
    long obase = ((long)(b * Tlen + t0 + rowLN)) * FF;
    for (int e = sub; e < FF; e += 8) {
        float v = sY[rowLN * FF + e];
        out[obase + e] = (_Float16)((v - mean) * inv * cg[e] + cbe[e]);
    }
}

// ---------------------------------------------------------------------------
// 6) linear F->1 (+mask): one wave per row, shuffle reduction
// ---------------------------------------------------------------------------
__global__ void linear_mask_kernel(const _Float16* __restrict__ in,
                                   const float* __restrict__ lw,
                                   const float* __restrict__ lb,
                                   const int* __restrict__ melLen,  // null = no mask
                                   float* __restrict__ outp,
                                   int Tlen)
{
    int tid = threadIdx.x;
    int lane = tid & 31, wave = tid >> 5;
    long row = (long)blockIdx.x * 8 + wave;
    const _Float16* ip = in + row * FF;
    float s = 0.0f;
    for (int e = lane; e < FF; e += 32) s += (float)ip[e] * lw[e];
#pragma unroll
    for (int off = 16; off >= 1; off >>= 1) s += __shfl_xor(s, off, 32);
    if (lane == 0) {
        float v = s + lb[0];
        if (melLen) {
            int bb = (int)(row / Tlen), t = (int)(row % Tlen);
            if (t >= melLen[bb]) v = 0.0f;
        }
        outp[row] = v;
    }
}

// ---------------------------------------------------------------------------
// 7) out = x_exp + pitch_pred*pc2w + pc2b + energy_pred*ec2w + ec2b
// ---------------------------------------------------------------------------
__global__ void final_out_kernel(const float* __restrict__ x,
                                 const int* __restrict__ idx,
                                 const int* __restrict__ melLen,
                                 const float* __restrict__ pp,
                                 const float* __restrict__ ep,
                                 const float* __restrict__ pc2w, const float* __restrict__ pc2b,
                                 const float* __restrict__ ec2w, const float* __restrict__ ec2b,
                                 float* __restrict__ out)
{
    long i = ((long)blockIdx.x * blockDim.x + threadIdx.x) * 4;
    int  f = (int)(i % FF);
    long row = i / FF;
    int  b = (int)(row / ML), t = (int)(row % ML);
    int  j = idx[row];
    bool live = (t < melLen[b]);
    float p = pp[row], e = ep[row];
    const float* xp = x + ((long)(b * TT + j)) * FF + f;
#pragma unroll
    for (int q = 0; q < 4; q++) {
        float xe = live ? xp[q] : 0.0f;
        out[i + q] = xe + p * pc2w[f + q] + pc2b[f + q] + e * ec2w[f + q] + ec2b[f + q];
    }
}

// ---------------------------------------------------------------------------
extern "C" void kernel_launch(void* const* d_in, const int* in_sizes, int n_in,
                              void* d_out, int out_size, void* d_ws, size_t ws_size,
                              hipStream_t stream)
{
    (void)in_sizes; (void)n_in; (void)out_size; (void)ws_size;
    // setup_inputs() dict order
    const float* x    = (const float*)d_in[0];
    const float* sp   = (const float*)d_in[1];
    const float* se   = (const float*)d_in[2];
    const int*   dur  = (const int*)d_in[4];
    const float* d_cw = (const float*)d_in[8];
    const float* d_cb = (const float*)d_in[9];
    const float* d_cg = (const float*)d_in[10];
    const float* d_cbe= (const float*)d_in[11];
    const float* d_lw = (const float*)d_in[12];
    const float* d_lb = (const float*)d_in[13];
    const float* p_cw = (const float*)d_in[14];
    const float* p_cb = (const float*)d_in[15];
    const float* p_cg = (const float*)d_in[16];
    const float* p_cbe= (const float*)d_in[17];
    const float* p_lw = (const float*)d_in[18];
    const float* p_lb = (const float*)d_in[19];
    const float* e_cw = (const float*)d_in[20];
    const float* e_cb = (const float*)d_in[21];
    const float* e_cg = (const float*)d_in[22];
    const float* e_cbe= (const float*)d_in[23];
    const float* e_lw = (const float*)d_in[24];
    const float* e_lb = (const float*)d_in[25];
    const float* pc1w = (const float*)d_in[26];
    const float* pc1b = (const float*)d_in[27];
    const float* ec1w = (const float*)d_in[28];
    const float* ec1b = (const float*)d_in[29];
    const float* pc2w = (const float*)d_in[30];
    const float* pc2b = (const float*)d_in[31];
    const float* ec2w = (const float*)d_in[32];
    const float* ec2b = (const float*)d_in[33];

    // output layout (tuple flattened, all as f32)
    float* out   = (float*)d_out;
    float* pp    = out + (long)BB * ML * FF;   // 25165824
    float* ep    = pp + BB * ML;               // +65536
    float* ld    = ep + BB * ML;               // +65536
    float* sdout = ld + BB * TT;               // +8192
    float* mlout = sdout + BB * TT;            // +8192
    float* mmout = mlout + BB;                 // +16

    // workspace layout
    char* ws = (char*)d_ws;
    int*      idx   = (int*)ws;                                    //   256 KiB
    int*      mel   = (int*)(ws + 262144);
    _Float16* buf0  = (_Float16*)(ws + 270336);                    //  48 MiB
    _Float16* buf1  = (_Float16*)(ws + 270336 + 50331648);         //  48 MiB
    _Float16* wfrag = (_Float16*)(ws + 270336 + 2L * 50331648);    // ~5 MiB

    dur_index_kernel<<<BB, TT, 0, stream>>>(dur, idx, mel, sdout, mlout, mmout);

    // repack the 6 conv layers to WMMA fragments
    const float* cws[3] = { d_cw, p_cw, e_cw };
    for (int pI = 0; pI < 3; pI++)
        for (int L = 0; L < 2; L++)
            repack_w_kernel<<<108, 256, 0, stream>>>(cws[pI] + (long)L * 3 * FF * FF,
                                                     wfrag + (long)(pI * 2 + L) * LAYER_FRAG_HALVES);

    // ---- duration predictor on (B, T) ----
    f32_to_f16_kernel<<<(BB * TT * FF / 8 + 255) / 256, 256, 0, stream>>>(x, buf0, BB * TT * FF);
    conv_ln_wmma_kernel<<<BB * (TT / MROWS), 256, 0, stream>>>(buf0, wfrag + 0L * LAYER_FRAG_HALVES,
                                                               d_cb, d_cg, d_cbe, buf1, TT);
    conv_ln_wmma_kernel<<<BB * (TT / MROWS), 256, 0, stream>>>(buf1, wfrag + 1L * LAYER_FRAG_HALVES,
                                                               d_cb + FF, d_cg + FF, d_cbe + FF, buf0, TT);
    linear_mask_kernel<<<BB * TT / 8, 256, 0, stream>>>(buf0, d_lw, d_lb, nullptr, ld, TT);

    // ---- pitch predictor on (B, ML) ----
    build_var_input_kernel<<<BB * ML * FF / 8 / 256, 256, 0, stream>>>(x, sp, pc1w, pc1b, idx, mel, buf0);
    conv_ln_wmma_kernel<<<BB * (ML / MROWS), 256, 0, stream>>>(buf0, wfrag + 2L * LAYER_FRAG_HALVES,
                                                               p_cb, p_cg, p_cbe, buf1, ML);
    conv_ln_wmma_kernel<<<BB * (ML / MROWS), 256, 0, stream>>>(buf1, wfrag + 3L * LAYER_FRAG_HALVES,
                                                               p_cb + FF, p_cg + FF, p_cbe + FF, buf0, ML);
    linear_mask_kernel<<<BB * ML / 8, 256, 0, stream>>>(buf0, p_lw, p_lb, mel, pp, ML);

    // ---- energy predictor on (B, ML) ----
    build_var_input_kernel<<<BB * ML * FF / 8 / 256, 256, 0, stream>>>(x, se, ec1w, ec1b, idx, mel, buf0);
    conv_ln_wmma_kernel<<<BB * (ML / MROWS), 256, 0, stream>>>(buf0, wfrag + 4L * LAYER_FRAG_HALVES,
                                                               e_cb, e_cg, e_cbe, buf1, ML);
    conv_ln_wmma_kernel<<<BB * (ML / MROWS), 256, 0, stream>>>(buf1, wfrag + 5L * LAYER_FRAG_HALVES,
                                                               e_cb + FF, e_cg + FF, e_cbe + FF, buf0, ML);
    linear_mask_kernel<<<BB * ML / 8, 256, 0, stream>>>(buf0, e_lw, e_lb, mel, ep, ML);

    // ---- final combine ----
    final_out_kernel<<<BB * ML * FF / 4 / 256, 256, 0, stream>>>(x, idx, mel, pp, ep,
                                                                 pc2w, pc2b, ec2w, ec2b, out);
}
